// CustomTaylorLayer_26182120636510
// MI455X (gfx1250) — compile-verified, matrix operand order checked by reading, not run on hardware
//
#include <hip/hip_runtime.h>
#include <hip/hip_bf16.h>

// Problem constants (from reference): B=2048, O=256, I=256, ORDER=5
#define BN 2048
#define ON 256
#define IN 256
#define NK 6              // ORDER+1 polynomial terms
#define KC (IN * NK)      // combined K dimension = 1536

typedef __attribute__((ext_vector_type(16))) _Float16 v16h;
typedef __attribute__((ext_vector_type(8)))  _Float16 v8h;
typedef __attribute__((ext_vector_type(8)))  float    v8f;

// ---------------------------------------------------------------------------
// Prep 1: P[b, k*IN + i] = x[b,i]^k   (f16), k = 0..5
// ---------------------------------------------------------------------------
__global__ void taylor_pow_kernel(const float* __restrict__ x,
                                  _Float16* __restrict__ P) {
    int idx = blockIdx.x * blockDim.x + threadIdx.x;   // over B*IN
    if (idx >= BN * IN) return;
    int b = idx >> 8;          // / IN
    int i = idx & (IN - 1);    // % IN
    float xv = x[idx];
    float p = 1.0f;
    _Float16* row = P + (size_t)b * KC + i;
#pragma unroll
    for (int k = 0; k < NK; ++k) {
        row[k * IN] = (_Float16)p;
        p *= xv;
    }
}

// ---------------------------------------------------------------------------
// Prep 2: W[o, k*IN + i] = coeffs[o,i,k] * weights[o,i]   (f16)
// coeffs layout: [O][I][NK] row-major
// ---------------------------------------------------------------------------
__global__ void taylor_wprep_kernel(const float* __restrict__ coeffs,
                                    const float* __restrict__ weights,
                                    _Float16* __restrict__ W) {
    int idx = blockIdx.x * blockDim.x + threadIdx.x;   // over O*IN
    if (idx >= ON * IN) return;
    int o = idx >> 8;
    int i = idx & (IN - 1);
    float wv = weights[idx];
    const float* c = coeffs + (size_t)idx * NK;
    _Float16* row = W + (size_t)o * KC + i;
#pragma unroll
    for (int k = 0; k < NK; ++k) {
        row[k * IN] = (_Float16)(c[k] * wv);
    }
}

// ---------------------------------------------------------------------------
// GEMM: out[b,o] = sum_kc P[b,kc] * W[o,kc]
// One wave32 per 32x32 output patch (2x2 register tile of 16x16 WMMA tiles).
// Per K-step of 32: 2 A fragments + 2 B fragments (8 global_load_b128) feed
// 4 independent v_wmma_f32_16x16x32_f16 accumulations -> 2 loads per WMMA
// and 4 parallel XDL dependency chains.
//
// Fragment layout (CDNA5 16-bit 16x32, wave32):
//   lane L (L<16):  row = L,    halves[0:8] = K[kk..kk+7],   halves[8:16] = K[kk+16..kk+23]
//   lane L (L>=16): row = L-16, halves[0:8] = K[kk+8..kk+15],halves[8:16] = K[kk+24..kk+31]
// Each half-fragment is a single aligned global_load_b128.
// ---------------------------------------------------------------------------
__global__ void __launch_bounds__(256)
taylor_wmma_gemm_kernel(const _Float16* __restrict__ P,
                        const _Float16* __restrict__ W,
                        float* __restrict__ out) {
    const int lane = threadIdx.x & 31;
    const int wave = threadIdx.x >> 5;               // 0..7
    const int tile = blockIdx.x * 8 + wave;          // 0..511
    const int mt = tile >> 3;                        // 32-row tile: 0..63
    const int nt = tile & 7;                         // 32-col tile: 0..7
    const int m0 = mt * 32;
    const int o0 = nt * 32;

    const int r    = lane & 15;                      // row within 16x16 tile
    const int half = lane >> 4;                      // 0 or 1

    const _Float16* arow0 = P + (size_t)(m0 + r) * KC + half * 8;
    const _Float16* arow1 = arow0 + (size_t)16 * KC;
    const _Float16* brow0 = W + (size_t)(o0 + r) * KC + half * 8;
    const _Float16* brow1 = brow0 + (size_t)16 * KC;

    v8f acc00 = {}, acc01 = {}, acc10 = {}, acc11 = {};

#pragma unroll 2
    for (int kk = 0; kk < KC; kk += 32) {
        v16h a0, a1, b0, b1;
        {
            v8h lo = *(const v8h*)(arow0 + kk);
            v8h hi = *(const v8h*)(arow0 + kk + 16);
#pragma unroll
            for (int j = 0; j < 8; ++j) { a0[j] = lo[j]; a0[j + 8] = hi[j]; }
        }
        {
            v8h lo = *(const v8h*)(arow1 + kk);
            v8h hi = *(const v8h*)(arow1 + kk + 16);
#pragma unroll
            for (int j = 0; j < 8; ++j) { a1[j] = lo[j]; a1[j + 8] = hi[j]; }
        }
        {
            v8h lo = *(const v8h*)(brow0 + kk);
            v8h hi = *(const v8h*)(brow0 + kk + 16);
#pragma unroll
            for (int j = 0; j < 8; ++j) { b0[j] = lo[j]; b0[j + 8] = hi[j]; }
        }
        {
            v8h lo = *(const v8h*)(brow1 + kk);
            v8h hi = *(const v8h*)(brow1 + kk + 16);
#pragma unroll
            for (int j = 0; j < 8; ++j) { b1[j] = lo[j]; b1[j + 8] = hi[j]; }
        }

        acc00 = __builtin_amdgcn_wmma_f32_16x16x32_f16(false, a0, false, b0,
                                                       (short)0, acc00, false, false);
        acc01 = __builtin_amdgcn_wmma_f32_16x16x32_f16(false, a0, false, b1,
                                                       (short)0, acc01, false, false);
        acc10 = __builtin_amdgcn_wmma_f32_16x16x32_f16(false, a1, false, b0,
                                                       (short)0, acc10, false, false);
        acc11 = __builtin_amdgcn_wmma_f32_16x16x32_f16(false, a1, false, b1,
                                                       (short)0, acc11, false, false);
    }

    // D layout per 16x16 tile: VGPR v, lanes 0-15 -> (M=v, N=lane),
    // lanes 16-31 -> (M=v+8, N=lane-16)
    float* o00 = out + (size_t)(m0 + half * 8) * ON + (o0 + r);
    float* o01 = o00 + 16;                 // +16 cols
    float* o10 = o00 + (size_t)16 * ON;    // +16 rows
    float* o11 = o10 + 16;
#pragma unroll
    for (int v = 0; v < 8; ++v) {
        o00[(size_t)v * ON] = acc00[v];
        o01[(size_t)v * ON] = acc01[v];
        o10[(size_t)v * ON] = acc10[v];
        o11[(size_t)v * ON] = acc11[v];
    }
}

// ---------------------------------------------------------------------------
extern "C" void kernel_launch(void* const* d_in, const int* in_sizes, int n_in,
                              void* d_out, int out_size, void* d_ws, size_t ws_size,
                              hipStream_t stream) {
    const float* x       = (const float*)d_in[0];   // [B, I]
    const float* coeffs  = (const float*)d_in[1];   // [O, I, NK]
    const float* weights = (const float*)d_in[2];   // [O, I]
    float* out = (float*)d_out;                     // [B, O]

    // Workspace layout (f16): P = B*KC (6 MB), W = O*KC (0.75 MB)
    _Float16* P = (_Float16*)d_ws;
    _Float16* W = (_Float16*)((char*)d_ws + (size_t)BN * KC * sizeof(_Float16));

    taylor_pow_kernel<<<(BN * IN) / 256, 256, 0, stream>>>(x, P);
    taylor_wprep_kernel<<<(ON * IN) / 256, 256, 0, stream>>>(coeffs, weights, W);

    // 512 patches of 32x32, 8 waves per block -> 64 blocks
    taylor_wmma_gemm_kernel<<<64, 256, 0, stream>>>(P, W, out);
}
